// MyTripletLoss_14688788152379
// MI455X (gfx1250) — matrix operand order repaired and checked by reading, not compiled, not run on hardware
//
#include <hip/hip_runtime.h>
#include <hip/hip_fp16.h>

#define NN 8192
#define DD 128
#define CC 10
#define MARGIN 0.0f
#define GAP 0.4f

typedef __attribute__((ext_vector_type(16))) _Float16 v16h;
typedef __attribute__((ext_vector_type(8)))  _Float16 v8h;
typedef __attribute__((ext_vector_type(8)))  float    v8f;

#if defined(__has_builtin)
#if __has_builtin(__builtin_amdgcn_global_load_async_to_lds_b128) && \
    __has_builtin(__builtin_amdgcn_s_wait_asynccnt)
#define USE_ASYNC_LDS 1
#endif
#endif

#if defined(USE_ASYNC_LDS)
typedef int v4i __attribute__((vector_size(16)));
typedef __attribute__((address_space(1))) v4i* v4i_gptr;   // global (AS1)
typedef __attribute__((address_space(3))) v4i* v4i_lptr;   // LDS (AS3)
#endif

// ---------------------------------------------------------------- fp32 -> f16
__global__ void k_cvt(const float* __restrict__ x, _Float16* __restrict__ xh) {
    int i = blockIdx.x * 256 + threadIdx.x;          // N*D threads
    xh[i] = (_Float16)x[i];
}

// ---------------------------------------------------------------- row sq-norms
__global__ void k_sq(const float* __restrict__ x, float* __restrict__ sq) {
    const int wave = threadIdx.x >> 5;
    const int lane = threadIdx.x & 31;
    const int row  = blockIdx.x * 4 + wave;          // 4 rows per 128-thr block
    const float4 v = *(const float4*)(x + (size_t)row * DD + lane * 4);
    float s = v.x * v.x + v.y * v.y + v.z * v.z + v.w * v.w;
#pragma unroll
    for (int off = 16; off > 0; off >>= 1) s += __shfl_xor(s, off, 32);
    if (lane == 0) sq[row] = s;
}

// ---------------------------------------------------------------- confusion flags
__global__ void k_confuse(const float* __restrict__ pred, const int* __restrict__ tgt,
                          float* __restrict__ conf) {
    int i = blockIdx.x * 256 + threadIdx.x;
    if (i >= NN) return;
    float l[CC];
    float m = -1e30f;
#pragma unroll
    for (int k = 0; k < CC; ++k) { l[k] = pred[i * CC + k]; m = fmaxf(m, l[k]); }
    float Z = 0.f;
#pragma unroll
    for (int k = 0; k < CC; ++k) { l[k] = __expf(l[k] - m); Z += l[k]; }
    float p1 = -1.f, p2 = -1.f; int i1 = 0;
#pragma unroll
    for (int k = 0; k < CC; ++k) {
        float v = l[k];
        if (v > p1) { p2 = p1; p1 = v; i1 = k; }
        else if (v > p2) { p2 = v; }
    }
    p1 /= Z; p2 /= Z;
    conf[i] = ((p1 - p2 <= GAP) || (i1 != tgt[i])) ? 1.0f : 0.0f;
}

// ---------------------------------------------------------------- fused Gram + hard mining
#define LDS_STRIDE 144   // halfs per LDS row (128 + 16 pad -> bank-friendly 32B frag reads)
#define JT 32            // columns per iteration (two 16x16 WMMA subtiles)
#define NT (NN / JT)     // 256 iterations

struct Stage { uint4 r[4]; };

// Issue the copy of tile rows j0..j0+31 of Xh toward LDS buffer `buf`.
__device__ __forceinline__ void stage_issue(const _Float16* __restrict__ xh, int j0,
                                            _Float16* buf, Stage& st) {
    int id = threadIdx.x;                       // 512 16B-chunks, 4 per thread
#pragma unroll
    for (int rep = 0; rep < 4; ++rep, id += 128) {
        const int row = id >> 4, c = id & 15;
        const _Float16* src = xh + (size_t)(j0 + row) * DD + c * 8;
#if defined(USE_ASYNC_LDS)
        _Float16* dst = buf + row * LDS_STRIDE + c * 8;
        __builtin_amdgcn_global_load_async_to_lds_b128(
            (v4i_gptr)src, (v4i_lptr)dst, 0, 0);
        (void)st;
#else
        st.r[rep] = *(const uint4*)src;
#endif
    }
}

// Complete the staged copy into `buf` (must precede the barrier).
__device__ __forceinline__ void stage_commit(_Float16* buf, Stage& st) {
#if defined(USE_ASYNC_LDS)
    __builtin_amdgcn_s_wait_asynccnt(0);
    (void)buf; (void)st;
#else
    int id = threadIdx.x;
#pragma unroll
    for (int rep = 0; rep < 4; ++rep, id += 128) {
        const int row = id >> 4, c = id & 15;
        *(uint4*)(buf + row * LDS_STRIDE + c * 8) = st.r[rep];
    }
#endif
}

__global__ void __launch_bounds__(128) k_mine(const _Float16* __restrict__ xh,
                                              const float* __restrict__ sq,
                                              const int* __restrict__ tgt,
                                              float* __restrict__ ap,
                                              float* __restrict__ an) {
    __shared__ __align__(16) _Float16 tile[2][JT * LDS_STRIDE];

    const int wave = threadIdx.x >> 5;
    const int lane = threadIdx.x & 31;
    const int half = lane >> 4;          // 0 or 1
    const int l16  = lane & 15;
    const int r0   = blockIdx.x * 64 + wave * 16;    // this wave's 16 rows

    // Preload A fragments for all of K=128 (4 steps of K=32), contiguous K order (matches B):
    // lane L holds row M=l16; halfs [0..15] = K = 32*kk + 16*half + (0..15)
    v16h a[4];
    const _Float16* arow = xh + (size_t)(r0 + l16) * DD;
#pragma unroll
    for (int kk = 0; kk < 4; ++kk) {
        a[kk] = *(const v16h*)(arow + kk * 32 + half * 16);
    }

    // C/D layout: VGPR v, lanes 0-15 -> row v, lanes 16-31 -> row v+8; col = l16
    float sqi[8]; int ti[8];
#pragma unroll
    for (int v = 0; v < 8; ++v) {
        const int r = r0 + v + 8 * half;
        sqi[v] = sq[r];
        ti[v]  = tgt[r];
    }
    float m_ap[8], m_an[8];
#pragma unroll
    for (int v = 0; v < 8; ++v) { m_ap[v] = -1e30f; m_an[v] = 1e30f; }

    // Prologue: stage tile 0, preload first column scalars
    Stage st;
    stage_issue(xh, 0, &tile[0][0], st);
    float sqj0 = sq[l16],      sqj1 = sq[16 + l16];
    int   tj0  = tgt[l16],     tj1  = tgt[16 + l16];
    stage_commit(&tile[0][0], st);
    __syncthreads();

    for (int jt = 0; jt < NT; ++jt) {
        const int p  = jt & 1;
        const int j0 = jt * JT;
        const bool more = (jt + 1 < NT);

        // Pipeline: issue next tile's copy + next column scalars now
        if (more) stage_issue(xh, j0 + JT, &tile[1 - p][0], st);
        float nsq0 = 0.f, nsq1 = 0.f; int ntj0 = 0, ntj1 = 0;
        if (more) {
            nsq0 = sq[j0 + JT + l16];      nsq1 = sq[j0 + JT + 16 + l16];
            ntj0 = tgt[j0 + JT + l16];     ntj1 = tgt[j0 + JT + 16 + l16];
        }

        // Two 16-wide column subtiles: B lane L holds col N = s*16 + l16,
        // K = 16*half + (0..15) -> one contiguous 32B chunk per fragment.
        v8f acc0 = {}, acc1 = {};
#pragma unroll
        for (int kk = 0; kk < 4; ++kk) {
            const v16h b0 = *(const v16h*)(&tile[p][l16 * LDS_STRIDE + kk * 32 + half * 16]);
            const v16h b1 = *(const v16h*)(&tile[p][(16 + l16) * LDS_STRIDE + kk * 32 + half * 16]);
            acc0 = __builtin_amdgcn_wmma_f32_16x16x32_f16(false, a[kk], false, b0, (short)0, acc0, false, false);
            acc1 = __builtin_amdgcn_wmma_f32_16x16x32_f16(false, a[kk], false, b1, (short)0, acc1, false, false);
        }

        // Epilogue: d^2 = |xi|^2 + |xj|^2 - 2*G ; masked hard mining
#pragma unroll
        for (int v = 0; v < 8; ++v) {
            const float d20 = sqi[v] + sqj0 - 2.0f * acc0[v];
            const float d21 = sqi[v] + sqj1 - 2.0f * acc1[v];
            const bool  s0  = (ti[v] == tj0);
            const bool  s1  = (ti[v] == tj1);
            m_ap[v] = (s0  && d20 > m_ap[v]) ? d20 : m_ap[v];
            m_an[v] = (!s0 && d20 < m_an[v]) ? d20 : m_an[v];
            m_ap[v] = (s1  && d21 > m_ap[v]) ? d21 : m_ap[v];
            m_an[v] = (!s1 && d21 < m_an[v]) ? d21 : m_an[v];
        }

        if (more) {
            stage_commit(&tile[1 - p][0], st);
            sqj0 = nsq0; sqj1 = nsq1; tj0 = ntj0; tj1 = ntj1;
        }
        __syncthreads();
    }

    // Reduce across the 16 lanes holding each row (xor<16 never crosses halves)
#pragma unroll
    for (int v = 0; v < 8; ++v) {
        float pa = m_ap[v], na = m_an[v];
#pragma unroll
        for (int off = 8; off >= 1; off >>= 1) {
            pa = fmaxf(pa, __shfl_xor(pa, off, 32));
            na = fminf(na, __shfl_xor(na, off, 32));
        }
        if (l16 == 0) {
            const int r = r0 + v + 8 * half;
            ap[r] = sqrtf(fmaxf(pa, 1e-12f));
            an[r] = sqrtf(fmaxf(na, 1e-12f));
        }
    }
}

// ---------------------------------------------------------------- final scalar
__global__ void k_final(const float* __restrict__ ap, const float* __restrict__ an,
                        const float* __restrict__ conf, float* __restrict__ out) {
    __shared__ float ssum[256];
    __shared__ float scnt[256];
    float s = 0.f, c = 0.f;
    for (int i = threadIdx.x; i < NN; i += 256) {
        const float f = conf[i];
        s += f * fmaxf(ap[i] - an[i] + MARGIN, 0.0f);
        c += f;
    }
    ssum[threadIdx.x] = s;
    scnt[threadIdx.x] = c;
    __syncthreads();
    for (int off = 128; off > 0; off >>= 1) {
        if (threadIdx.x < off) {
            ssum[threadIdx.x] += ssum[threadIdx.x + off];
            scnt[threadIdx.x] += scnt[threadIdx.x + off];
        }
        __syncthreads();
    }
    if (threadIdx.x == 0) {
        const float cnt = scnt[0];
        out[0] = (cnt > 0.f) ? (ssum[0] / fmaxf(cnt, 1.f)) : 0.f;
    }
}

// ---------------------------------------------------------------- launch
extern "C" void kernel_launch(void* const* d_in, const int* in_sizes, int n_in,
                              void* d_out, int out_size, void* d_ws, size_t ws_size,
                              hipStream_t stream) {
    const float* x    = (const float*)d_in[0];
    const float* pred = (const float*)d_in[1];
    const int*   tgt  = (const int*)d_in[2];
    float*       out  = (float*)d_out;

    char* ws = (char*)d_ws;
    _Float16* xh = (_Float16*)ws;  ws += (size_t)NN * DD * sizeof(_Float16);
    float* sq    = (float*)ws;     ws += (size_t)NN * sizeof(float);
    float* ap    = (float*)ws;     ws += (size_t)NN * sizeof(float);
    float* an    = (float*)ws;     ws += (size_t)NN * sizeof(float);
    float* conf  = (float*)ws;     ws += (size_t)NN * sizeof(float);

    k_cvt    <<<(NN * DD) / 256, 256, 0, stream>>>(x, xh);
    k_sq     <<<NN / 4,          128, 0, stream>>>(x, sq);
    k_confuse<<<NN / 256,        256, 0, stream>>>(pred, tgt, conf);
    k_mine   <<<NN / 64,         128, 0, stream>>>(xh, sq, tgt, ap, an);
    k_final  <<<1,               256, 0, stream>>>(ap, an, conf, out);
}
